// ATTENTION_5549097746558
// MI455X (gfx1250) — compile-verified, matrix operand order checked by reading, not run on hardware
//
#include <hip/hip_runtime.h>

// ---------------------------------------------------------------------------
// TiSASRec-style 2-block transformer forward for MI455X (gfx1250, wave32).
// All dense matmuls use v_wmma_f32_16x16x32_bf16 (f32 accumulate).
// Time-relative (data-dependent-gather) terms use VALU against L2-hot tables.
// Intermediates live in d_ws; at these sizes everything is L2-resident.
// ---------------------------------------------------------------------------

#define Bc       16
#define Lc       256
#define Hc       64
#define NHc      2
#define HSc      32
#define NBc      2
#define Mc       (Bc * Lc)         // 4096 rows
#define TROWS    257               // TIME_SPAN + 1
#define ITEMNUM  50000
#define NEGBIG   (-4294967295.0f)

typedef __bf16 bf16_t;
typedef __attribute__((ext_vector_type(16))) __bf16 v16bf;
typedef __attribute__((ext_vector_type(8)))  float  v8f;

static __device__ __forceinline__ bf16_t f2bf(float f) {
  unsigned u = __builtin_bit_cast(unsigned, f);
  unsigned r = (u + 0x7FFFu + ((u >> 16) & 1u)) >> 16;   // RNE
  unsigned short s = (unsigned short)r;
  return __builtin_bit_cast(bf16_t, s);
}
static __device__ __forceinline__ float bf2f(bf16_t b) {
  unsigned short s = __builtin_bit_cast(unsigned short, b);
  unsigned u = ((unsigned)s) << 16;
  return __builtin_bit_cast(float, u);
}

static __device__ __forceinline__ v8f wmma_bf16(v16bf a, v16bf b, v8f c) {
  // D = A(16x32) * B(32x16) + C, f32 accumulate
  return __builtin_amdgcn_wmma_f32_16x16x32_bf16(false, a, false, b,
                                                 (short)0, c, false, false);
}

// A-operand loader: src is a [16 x >=32] row-major tile (rows = M for A, or
// rows = N when the B-operand source is stored [N][K], e.g. K / weight tiles).
// Per-lane packing per ISA 7.12.2 (16-bit A 16x32).
static __device__ __forceinline__ v16bf load_frag_nk(const bf16_t* src, int ld) {
  int lane = threadIdx.x & 31;
  int r = lane & 15, half = lane >> 4;
  v16bf f;
#pragma unroll
  for (int v = 0; v < 8; ++v) {
    int kb = ((v < 4) ? (v * 2) : (16 + (v - 4) * 2)) + half * 8;
    f[2 * v]     = src[r * ld + kb];
    f[2 * v + 1] = src[r * ld + kb + 1];
  }
  return f;
}
// B-operand loader when source is stored [K rows x N cols] row-major.
static __device__ __forceinline__ v16bf load_frag_kn(const bf16_t* src, int ld) {
  int lane = threadIdx.x & 31;
  int n = lane & 15, half = lane >> 4;
  v16bf f;
#pragma unroll
  for (int v = 0; v < 8; ++v) {
    int kb = ((v < 4) ? (v * 2) : (16 + (v - 4) * 2)) + half * 8;
    f[2 * v]     = src[kb * ld + n];
    f[2 * v + 1] = src[(kb + 1) * ld + n];
  }
  return f;
}

// ---------------------------------------------------------------------------
// Prep: mask padded rows of seqs, build abs-pos K (bf16) / V (f32) embeddings.
__global__ __launch_bounds__(256) void k_prep(
    const int* __restrict__ sdata, const float* __restrict__ seqs,
    const int* __restrict__ positions, const float* __restrict__ apK,
    const float* __restrict__ apV, float* __restrict__ seqs_cur,
    bf16_t* __restrict__ aKbf, float* __restrict__ aVf) {
  int idx = blockIdx.x * 256 + threadIdx.x;
  if (idx >= Mc * Hc) return;
  int bl = idx >> 6, hd = idx & 63;
  float keep = (sdata[bl] == ITEMNUM - 1) ? 0.f : 1.f;
  seqs_cur[idx] = seqs[idx] * keep;
  int p = positions[bl];
  float pk = (p != 0) ? 1.f : 0.f;
  aKbf[idx] = f2bf(apK[p * Hc + hd] * pk);
  aVf[idx]  = apV[p * Hc + hd] * pk;
}

__global__ __launch_bounds__(256) void k_cvt(const float* __restrict__ src,
                                             bf16_t* __restrict__ dst, int n) {
  int i = blockIdx.x * 256 + threadIdx.x;
  if (i < n) dst[i] = f2bf(src[i]);
}

__global__ __launch_bounds__(256) void k_vplus(const float* __restrict__ Vf,
                                               const float* __restrict__ aVf,
                                               bf16_t* __restrict__ Vp, int n) {
  int i = blockIdx.x * 256 + threadIdx.x;
  if (i < n) Vp[i] = f2bf(Vf[i] + aVf[i]);
}

// ---------------------------------------------------------------------------
// Row layernorm over H=64: one wave32 per row, 2 elems/lane, shfl reductions.
__global__ __launch_bounds__(256) void k_layernorm(
    const float* __restrict__ X, const float* __restrict__ g,
    const float* __restrict__ b, float* __restrict__ Y, int M) {
  int w = threadIdx.x >> 5, lane = threadIdx.x & 31;
  int row = blockIdx.x * 8 + w;
  if (row >= M) return;
  float x0 = X[row * 64 + lane], x1 = X[row * 64 + 32 + lane];
  float s = x0 + x1;
#pragma unroll
  for (int o = 16; o; o >>= 1) s += __shfl_xor(s, o, 32);
  float mean = s * (1.f / 64.f);
  float d0 = x0 - mean, d1 = x1 - mean;
  float vs = d0 * d0 + d1 * d1;
#pragma unroll
  for (int o = 16; o; o >>= 1) vs += __shfl_xor(vs, o, 32);
  float inv = rsqrtf(vs * (1.f / 64.f) + 1e-8f);
  Y[row * 64 + lane]      = g[lane] * d0 * inv + b[lane];
  Y[row * 64 + 32 + lane] = g[lane + 32] * d1 * inv + b[lane + 32];
}

// ---------------------------------------------------------------------------
// Y[M,64] = X[M,64] @ W^T + bias  (W stored [N=64][K=64], bf16)
// optional relu, residual add, pad-row zeroing, bf16 mirror output.
// 256 thr = 8 waves; each wave owns one 16x16 tile of a 32x64 block.
__global__ __launch_bounds__(256) void k_gemm64(
    const float* __restrict__ X, const bf16_t* __restrict__ W,
    const float* __restrict__ bias, float* __restrict__ Y,
    bf16_t* __restrict__ Ybf, const float* __restrict__ resid,
    const int* __restrict__ maskdata, int relu, int M) {
  __shared__ bf16_t sX[32 * 64];
  __shared__ bf16_t sW[64 * 64];
  int tid = threadIdx.x;
  int row0 = blockIdx.x * 32;
  if (row0 + 32 < M)  // emits global_prefetch_b8 for next M-tile
    __builtin_prefetch(&X[(size_t)(row0 + 32) * 64 + tid], 0, 1);
#pragma unroll
  for (int i = tid; i < 32 * 64; i += 256)
    sX[i] = f2bf(X[(size_t)(row0 + (i >> 6)) * 64 + (i & 63)]);
#pragma unroll
  for (int i = tid; i < 64 * 64; i += 256) sW[i] = W[i];
  __syncthreads();

  int w = tid >> 5;
  int mt = w & 1, nt = w >> 1;            // 2 M-tiles x 4 N-tiles
  v8f acc = {};
#pragma unroll
  for (int kk = 0; kk < 64; kk += 32) {
    v16bf a = load_frag_nk(sX + (mt * 16) * 64 + kk, 64);
    v16bf bfr = load_frag_nk(sW + (nt * 16) * 64 + kk, 64);
    acc = wmma_bf16(a, bfr, acc);
  }
  int lane = tid & 31;
  int n = nt * 16 + (lane & 15);
  int mb = mt * 16 + (lane >> 4) * 8;
  float bn = bias[n];
#pragma unroll
  for (int v = 0; v < 8; ++v) {
    int m = row0 + mb + v;
    float val = acc[v] + bn;
    if (relu) val = fmaxf(val, 0.f);
    if (resid) val += resid[(size_t)m * 64 + n];
    if (maskdata && maskdata[m] == ITEMNUM - 1) val = 0.f;
    Y[(size_t)m * 64 + n] = val;
    if (Ybf) Ybf[(size_t)m * 64 + n] = f2bf(val);
  }
}

// ---------------------------------------------------------------------------
// Fused attention for one (b, head, 16-row l-tile). 256 thr = 8 waves.
//   scores = WMMA(Q,K^T) + WMMA(Q,aK^T) + VALU time-K term, scale, mask
//   softmax, out = WMMA(A, V+aV) + VALU time-V term, + q_in residual
__global__ __launch_bounds__(256) void k_attn(
    const float* __restrict__ Qf, const bf16_t* __restrict__ Kbf,
    const bf16_t* __restrict__ aKbf, const bf16_t* __restrict__ Vpbf,
    const float* __restrict__ q_in, const int* __restrict__ sdata,
    const int* __restrict__ tmat, const bf16_t* __restrict__ TKbf,
    const bf16_t* __restrict__ TVbf, float* __restrict__ seqs_out) {
  __shared__ float          sQ[16 * 32];    // Q head slice (f32, for time dots)
  __shared__ bf16_t         sQb[16 * 32];   // Q head slice (bf16, WMMA A)
  __shared__ float          sS[16 * 256];   // scores -> probs (f32)
  __shared__ bf16_t         sAb[16 * 256];  // probs (bf16, WMMA A)
  __shared__ bf16_t         sVp[256 * 32];  // (V+aV) head slice [m][d]
  __shared__ unsigned short sTm[16 * 256];  // time indices (<= 256 fits u16)
  __shared__ float          sO[16 * 32];    // WMMA A@Vplus result

  int ltile = blockIdx.x, h = blockIdx.y, b = blockIdx.z;
  int tid = threadIdx.x, w = tid >> 5, lane = tid & 31;
  int l0 = ltile * 16;

#pragma unroll
  for (int i = tid; i < 16 * 32; i += 256) {
    int l = i >> 5, d = i & 31;
    float q = Qf[(size_t)((b * Lc + l0 + l) * 64) + h * 32 + d];
    sQ[i] = q;
    sQb[i] = f2bf(q);
  }
#pragma unroll
  for (int i = tid; i < 16 * 256; i += 256) {
    int l = i >> 8, m = i & 255;
    sTm[i] = (unsigned short)tmat[(size_t)(b * Lc + l0 + l) * Lc + m];
  }
#pragma unroll
  for (int i = tid; i < 256 * 32; i += 256) {
    int m = i >> 5, d = i & 31;
    sVp[i] = Vpbf[(size_t)(b * Lc + m) * 64 + h * 32 + d];
  }
  __syncthreads();

  // ---- scores: 16 m-tiles split over 8 waves; K-dim == HS == 32 -> 1 WMMA each
  v16bf aQ = load_frag_nk(sQb, 32);
  for (int mt = w; mt < 16; mt += 8) {
    const bf16_t* Ksrc = Kbf  + (size_t)(b * Lc + mt * 16) * 64 + h * 32;
    const bf16_t* Asrc = aKbf + (size_t)(b * Lc + mt * 16) * 64 + h * 32;
    v16bf bK = load_frag_nk(Ksrc, 64);   // rows = m (N), cols = d (K)
    v16bf bA = load_frag_nk(Asrc, 64);
    v8f acc = {};
    acc = wmma_bf16(aQ, bK, acc);
    acc = wmma_bf16(aQ, bA, acc);
    int n = lane & 15, rb = (lane >> 4) * 8;
#pragma unroll
    for (int v = 0; v < 8; ++v)
      sS[(rb + v) * 256 + mt * 16 + n] = acc[v];
  }
  __syncthreads();

  // ---- time-K term (data-dependent gather), scale, mask
  const float scale = 0.17677669529663688f;  // 1/sqrt(32)
  for (int i = tid; i < 16 * 256; i += 256) {
    int l = i >> 8, m = i & 255;
    int tt = sTm[i];
    const bf16_t* tr = TKbf + (size_t)tt * 64 + h * 32;
    float acc = 0.f;
#pragma unroll 8
    for (int d = 0; d < 32; ++d) acc += sQ[l * 32 + d] * bf2f(tr[d]);
    float val = (sS[i] + acc) * scale;
    int Lg = l0 + l;
    bool masked = (m > Lg) || (sdata[b * Lc + m] == ITEMNUM - 1);
    sS[i] = masked ? NEGBIG : val;
  }
  __syncthreads();

  // ---- softmax: 2 rows per wave, 8 cols per lane
  for (int r = w * 2; r < w * 2 + 2; ++r) {
    float* row = sS + r * 256;
    float mx = -INFINITY;
#pragma unroll
    for (int j = 0; j < 8; ++j) mx = fmaxf(mx, row[lane + j * 32]);
#pragma unroll
    for (int o = 16; o; o >>= 1) mx = fmaxf(mx, __shfl_xor(mx, o, 32));
    float vals[8], sum = 0.f;
#pragma unroll
    for (int j = 0; j < 8; ++j) {
      float e = __expf(row[lane + j * 32] - mx);
      vals[j] = e; sum += e;
    }
#pragma unroll
    for (int o = 16; o; o >>= 1) sum += __shfl_xor(sum, o, 32);
    float rinv = 1.f / sum;
#pragma unroll
    for (int j = 0; j < 8; ++j) {
      float p = vals[j] * rinv;
      row[lane + j * 32] = p;
      sAb[r * 256 + lane + j * 32] = f2bf(p);
    }
  }
  __syncthreads();

  // ---- out = A @ (V+aV): waves 0,1 each own a 16x16 d-tile; K=256 -> 8 WMMAs
  if (w < 2) {
    v8f acc = {};
#pragma unroll
    for (int kk = 0; kk < 8; ++kk) {
      v16bf aA = load_frag_nk(sAb + kk * 32, 256);
      v16bf bV = load_frag_kn(sVp + (kk * 32) * 32 + w * 16, 32);
      acc = wmma_bf16(aA, bV, acc);
    }
    int n = w * 16 + (lane & 15), rb = (lane >> 4) * 8;
#pragma unroll
    for (int v = 0; v < 8; ++v) sO[(rb + v) * 32 + n] = acc[v];
  }
  __syncthreads();

  // ---- time-V term + residual, write back
  for (int i = tid; i < 16 * 32; i += 256) {
    int l = i >> 5, d = i & 31;
    const float* prow = sS + l * 256;
    const unsigned short* trow = sTm + l * 256;
    float acc = 0.f;
    for (int m = 0; m < 256; ++m)
      acc += prow[m] * bf2f(TVbf[(size_t)trow[m] * 64 + h * 32 + d]);
    int Lg = l0 + l;
    size_t off = (size_t)(b * Lc + Lg) * 64 + h * 32 + d;
    seqs_out[off] = q_in[off] + sO[i] + acc;
  }
}

// ---------------------------------------------------------------------------
extern "C" void kernel_launch(void* const* d_in, const int* in_sizes, int n_in,
                              void* d_out, int out_size, void* d_ws,
                              size_t ws_size, hipStream_t stream) {
  (void)in_sizes; (void)n_in; (void)out_size; (void)ws_size;
  const int*   sdata     = (const int*)d_in[0];
  const float* seqs0     = (const float*)d_in[1];
  const int*   positions = (const int*)d_in[2];
  const int*   tmat      = (const int*)d_in[3];
  const float* apK       = (const float*)d_in[4];
  const float* apV       = (const float*)d_in[5];
  const float* tK        = (const float*)d_in[6];
  const float* tV        = (const float*)d_in[7];
  const float* Qw        = (const float*)d_in[8];
  const float* Qb        = (const float*)d_in[9];
  const float* Kw        = (const float*)d_in[10];
  const float* Kb        = (const float*)d_in[11];
  const float* Vw        = (const float*)d_in[12];
  const float* Vb        = (const float*)d_in[13];
  const float* ln1g      = (const float*)d_in[14];
  const float* ln1b      = (const float*)d_in[15];
  const float* ln2g      = (const float*)d_in[16];
  const float* ln2b      = (const float*)d_in[17];
  const float* W1        = (const float*)d_in[18];
  const float* b1        = (const float*)d_in[19];
  const float* W2        = (const float*)d_in[20];
  const float* b2        = (const float*)d_in[21];
  const float* lastg     = (const float*)d_in[22];
  const float* lastb     = (const float*)d_in[23];

  char* cur = (char*)d_ws;
  auto alloc = [&](size_t bytes) -> void* {
    void* r = (void*)cur;
    cur += (bytes + 255) & ~(size_t)255;
    return r;
  };
  const size_t MH = (size_t)Mc * Hc;          // 262144 elems
  float*  seqs_cur = (float*)alloc(MH * 4);
  float*  q_in     = (float*)alloc(MH * 4);
  float*  Qf       = (float*)alloc(MH * 4);
  float*  Kf       = (float*)alloc(MH * 4);
  float*  Vf       = (float*)alloc(MH * 4);
  float*  aVf      = (float*)alloc(MH * 4);
  float*  h1       = (float*)alloc(MH * 4);
  bf16_t* Qbf      = (bf16_t*)alloc(MH * 2);
  bf16_t* Kbf      = (bf16_t*)alloc(MH * 2);
  bf16_t* aKbf     = (bf16_t*)alloc(MH * 2);
  bf16_t* Vpbf     = (bf16_t*)alloc(MH * 2);
  bf16_t* Qwbf     = (bf16_t*)alloc((size_t)NBc * 64 * 64 * 2);
  bf16_t* Kwbf     = (bf16_t*)alloc((size_t)NBc * 64 * 64 * 2);
  bf16_t* Vwbf     = (bf16_t*)alloc((size_t)NBc * 64 * 64 * 2);
  bf16_t* W1bf     = (bf16_t*)alloc((size_t)NBc * 64 * 64 * 2);
  bf16_t* W2bf     = (bf16_t*)alloc((size_t)NBc * 64 * 64 * 2);
  bf16_t* TKbf     = (bf16_t*)alloc((size_t)TROWS * 64 * 2);
  bf16_t* TVbf     = (bf16_t*)alloc((size_t)TROWS * 64 * 2);

  const int NW = NBc * 64 * 64;               // 8192
  const int NT = TROWS * 64;                  // 16448

  k_prep<<<(MH + 255) / 256, 256, 0, stream>>>(sdata, seqs0, positions, apK,
                                               apV, seqs_cur, aKbf, aVf);
  k_cvt<<<(NW + 255) / 256, 256, 0, stream>>>(Qw, Qwbf, NW);
  k_cvt<<<(NW + 255) / 256, 256, 0, stream>>>(Kw, Kwbf, NW);
  k_cvt<<<(NW + 255) / 256, 256, 0, stream>>>(Vw, Vwbf, NW);
  k_cvt<<<(NW + 255) / 256, 256, 0, stream>>>(W1, W1bf, NW);
  k_cvt<<<(NW + 255) / 256, 256, 0, stream>>>(W2, W2bf, NW);
  k_cvt<<<(NT + 255) / 256, 256, 0, stream>>>(tK, TKbf, NT);
  k_cvt<<<(NT + 255) / 256, 256, 0, stream>>>(tV, TVbf, NT);

  for (int i = 0; i < NBc; ++i) {
    const size_t wo = (size_t)i * 64 * 64;
    const size_t vo = (size_t)i * 64;
    // q_in = LN1(seqs)
    k_layernorm<<<Mc / 8, 256, 0, stream>>>(seqs_cur, ln1g + vo, ln1b + vo,
                                            q_in, Mc);
    // Q = q_in @ Qw^T + Qb ; K,V from (un-normed) seqs
    k_gemm64<<<Mc / 32, 256, 0, stream>>>(q_in, Qwbf + wo, Qb + vo, Qf, Qbf,
                                          nullptr, nullptr, 0, Mc);
    k_gemm64<<<Mc / 32, 256, 0, stream>>>(seqs_cur, Kwbf + wo, Kb + vo, Kf,
                                          Kbf, nullptr, nullptr, 0, Mc);
    k_gemm64<<<Mc / 32, 256, 0, stream>>>(seqs_cur, Vwbf + wo, Vb + vo, Vf,
                                          nullptr, nullptr, nullptr, 0, Mc);
    // V+aV fused operand for the output GEMM
    k_vplus<<<(MH + 255) / 256, 256, 0, stream>>>(Vf, aVf, Vpbf, (int)MH);
    // fused attention; writes seqs = q_in + attn_out
    k_attn<<<dim3(Lc / 16, NHc, Bc), 256, 0, stream>>>(
        Qf, Kbf, aKbf, Vpbf, q_in, sdata, tmat, TKbf, TVbf, seqs_cur);
    // LN2 (in place), FFN with relu, then FFN2 + residual + pad-row mask
    k_layernorm<<<Mc / 8, 256, 0, stream>>>(seqs_cur, ln2g + vo, ln2b + vo,
                                            seqs_cur, Mc);
    k_gemm64<<<Mc / 32, 256, 0, stream>>>(seqs_cur, W1bf + wo, b1 + vo, h1,
                                          nullptr, nullptr, nullptr, 1, Mc);
    k_gemm64<<<Mc / 32, 256, 0, stream>>>(h1, W2bf + wo, b2 + vo, seqs_cur,
                                          nullptr, seqs_cur, sdata, 0, Mc);
  }
  // final layernorm -> d_out (f32)
  k_layernorm<<<Mc / 8, 256, 0, stream>>>(seqs_cur, lastg, lastb,
                                          (float*)d_out, Mc);
}
